// mymodel_19327352832008
// MI455X (gfx1250) — compile-verified, matrix operand order checked by reading, not run on hardware
//
#include <hip/hip_runtime.h>
#include <hip/hip_bf16.h>

typedef __bf16 bf16_t;
typedef __attribute__((ext_vector_type(16))) bf16_t v16bf;
typedef __attribute__((ext_vector_type(8)))  bf16_t v8bf;
typedef __attribute__((ext_vector_type(8)))  float  v8f;
typedef __attribute__((ext_vector_type(4)))  float  v4f;

#define NN    8192   // rows of A / nodes
#define KK    8192   // inner dim (== NN, A is square)
#define NCOL  128    // concat(F, new_features) width
#define DHALF 64     // ReLU applies to columns [0, 64)

// ---------------------------------------------------------------------------
// Pack B0^T [128 x 8192] bf16 from F [8192 x 64] and G [8192 x 64] (fp32).
// B0t[n][k] = (n < 64) ? F[k][n] : G[k][n-64]
// ---------------------------------------------------------------------------
__global__ __launch_bounds__(256) void pack_b0_kernel(
    const float* __restrict__ F, const float* __restrict__ G,
    bf16_t* __restrict__ B0t)
{
    int idx = blockIdx.x * 256 + threadIdx.x;      // 0 .. 128*8192-1
    int k = idx & (KK - 1);                        // fast-moving -> coalesced store
    int n = idx >> 13;
    float v = (n < DHALF) ? F[(size_t)k * DHALF + n]
                          : G[(size_t)k * DHALF + (n - DHALF)];
    B0t[(size_t)n * KK + k] = (bf16_t)v;
}

// ---------------------------------------------------------------------------
// C = A @ B with A fp32 [8192 x 8192] row-major (converted to bf16 in-reg)
// and B given transposed as Bt bf16 [128 x 8192].
// mode 0: write bf16 transposed outT[128 x 8192], ReLU on n<64  (layer-1)
// mode 1: write fp32 row-major out[8192 x 128],   ReLU on n<64  (layer-2)
//
// Block = 256 threads = 8 wave32. Wave w: rows (blk*64 + (w>>1)*16),
// columns ((w&1)*64 .. +64) as 4 WMMA tiles. Wave pairs share A rows (WGP$).
// ---------------------------------------------------------------------------
__global__ __launch_bounds__(256, 2) void gemm_bf16_wmma_kernel(
    const float*  __restrict__ A,
    const bf16_t* __restrict__ Bt,
    bf16_t*       __restrict__ outT,
    float*        __restrict__ out,
    int mode)
{
    const int wave  = threadIdx.x >> 5;
    const int lane  = threadIdx.x & 31;
    const int lrow  = lane & 15;              // M (A frag) / N (B frag, C tile)
    const int khalf = (lane >> 4) << 3;       // 0 or 8: K phase per 16-bit layout

    const int rowblk = blockIdx.x * 64 + (wave >> 1) * 16;
    const int nbase  = (wave & 1) * 64;

    const float*  ap  = A  + (size_t)(rowblk + lrow) * KK + khalf;
    const bf16_t* bp0 = Bt + (size_t)(nbase  + lrow) * KK + khalf;

    v8f acc[4] = {};   // f32 accumulators, 4 tiles of 16x16

    #pragma unroll 2
    for (int k0 = 0; k0 < KK; k0 += 32) {
        // --- A fragment: 16 fp32 -> v16bf per the 16-bit 16x32 A layout ---
        // lane<16 : elements 0-7 = K k0..k0+7,   8-15 = K k0+16..k0+23
        // lane>=16: elements 0-7 = K k0+8..+15,  8-15 = K k0+24..+31
        v4f f0 = *(const v4f*)(ap + k0);
        v4f f1 = *(const v4f*)(ap + k0 + 4);
        v4f f2 = *(const v4f*)(ap + k0 + 16);
        v4f f3 = *(const v4f*)(ap + k0 + 20);

        // prefetch A ~8 k-iterations (1 KB) ahead into cache hierarchy
        __builtin_prefetch(ap + k0 + 256, 0, 0);

        v16bf a;
        #pragma unroll
        for (int i = 0; i < 4; ++i) {
            a[i]      = (bf16_t)f0[i];
            a[4 + i]  = (bf16_t)f1[i];
            a[8 + i]  = (bf16_t)f2[i];
            a[12 + i] = (bf16_t)f3[i];
        }

        // --- 4 column tiles: B frag is contiguous bf16 from Bt (same K map) ---
        #pragma unroll
        for (int t = 0; t < 4; ++t) {
            const bf16_t* bp = bp0 + (size_t)t * 16 * KK + k0;
            v8bf b0 = *(const v8bf*)(bp);
            v8bf b1 = *(const v8bf*)(bp + 16);
            v16bf b = __builtin_shufflevector(b0, b1,
                       0,1,2,3,4,5,6,7,8,9,10,11,12,13,14,15);
            acc[t] = __builtin_amdgcn_wmma_f32_16x16x32_bf16(
                         false, a, false, b, (short)0, acc[t], false, false);
        }
    }

    // --- Store: C/D layout: VGPR r, lanes 0-15 -> M=r, lanes 16-31 -> M=8+r;
    //            N = lane & 15 within the tile.
    const int mbase = rowblk + ((lane >> 4) << 3);
    const bool relu = (nbase == 0);           // columns [0,64) get ReLU

    if (mode == 0) {
        #pragma unroll
        for (int t = 0; t < 4; ++t) {
            int n = nbase + t * 16 + lrow;
            bf16_t* op = outT + (size_t)n * KK + mbase;
            #pragma unroll
            for (int r = 0; r < 8; ++r) {
                float v = acc[t][r];
                if (relu) v = fmaxf(v, 0.0f);
                op[r] = (bf16_t)v;
            }
        }
    } else {
        #pragma unroll
        for (int t = 0; t < 4; ++t) {
            int n = nbase + t * 16 + lrow;
            float* op = out + (size_t)mbase * NCOL + n;
            #pragma unroll
            for (int r = 0; r < 8; ++r) {
                float v = acc[t][r];
                if (relu) v = fmaxf(v, 0.0f);
                op[(size_t)r * NCOL] = v;
            }
        }
    }
}

// ---------------------------------------------------------------------------
// Launch: pack -> GEMM1 (A@B0 -> B1^T bf16, relu half) -> GEMM2 (A@B1 -> out)
// ---------------------------------------------------------------------------
extern "C" void kernel_launch(void* const* d_in, const int* in_sizes, int n_in,
                              void* d_out, int out_size, void* d_ws, size_t ws_size,
                              hipStream_t stream) {
    (void)in_sizes; (void)n_in; (void)out_size; (void)ws_size;

    const float* A = (const float*)d_in[0];   // [8192 x 8192]
    const float* F = (const float*)d_in[1];   // [8192 x 64]
    const float* G = (const float*)d_in[2];   // [8192 x 64]
    float* out = (float*)d_out;               // [8192 x 128]

    bf16_t* B0t = (bf16_t*)d_ws;                      // 128*8192*2 = 2 MB
    bf16_t* B1t = B0t + (size_t)NCOL * KK;            // next 2 MB

    pack_b0_kernel<<<(NCOL * KK) / 256, 256, 0, stream>>>(F, G, B0t);

    gemm_bf16_wmma_kernel<<<NN / 64, 256, 0, stream>>>(A, B0t, B1t, nullptr, 0);
    gemm_bf16_wmma_kernel<<<NN / 64, 256, 0, stream>>>(A, B1t, nullptr, out, 1);
}
